// ElectronicSchnet_45801531245176
// MI455X (gfx1250) — compile-verified, matrix operand order checked by reading, not run on hardware
//
#include <hip/hip_runtime.h>
#include <hip/hip_bf16.h>

// ---------------- problem constants ----------------
#define BS    256
#define NE    64     // n_elec
#define NN    16     // n_nuc
#define NP    80     // n_part
#define NB    32     // basis
#define KD    128    // kernel dim
#define DD    128    // embedding dim
#define NL    3

typedef __bf16 bf16_t;
typedef __attribute__((ext_vector_type(16))) __bf16 v16bf;
typedef __attribute__((ext_vector_type(8)))  __bf16 v8bf;
typedef __attribute__((ext_vector_type(8)))  float  v8f;

#define V8F_ZERO {0.f,0.f,0.f,0.f,0.f,0.f,0.f,0.f}

// shifted softplus via HW transcendentals: max(x,0)+log(1+exp(-|x|)) - log 2
__device__ __forceinline__ float sspf(float x) {
    float e = __expf(-fabsf(x));                 // v_exp_f32
    return fmaxf(x, 0.f) + __logf(1.f + e) - 0.69314718055994531f;  // v_log_f32
}

// WMMA bf16 16x16x32, f32 accumulate
__device__ __forceinline__ v8f wmma_bf16(v16bf a, v16bf b, v8f c) {
    return __builtin_amdgcn_wmma_f32_16x16x32_bf16(
        /*neg_a=*/false, a, /*neg_b=*/false, b,
        /*c_mod=*/(short)0, c, /*reuse_a=*/false, /*reuse_b=*/false);
}

// gfx1250 async global->LDS copy of 16B, tracked by ASYNCcnt.
// lds_addr: workgroup-relative LDS byte address (low 32 bits of generic ptr).
__device__ __forceinline__ void async_copy_b128(unsigned lds_addr, const void* gptr) {
    asm volatile("global_load_async_to_lds_b128 %0, %1, off"
                 :: "v"(lds_addr), "v"((unsigned long long)(uintptr_t)gptr)
                 : "memory");
}
__device__ __forceinline__ unsigned lds_off(const void* p) {
    return (unsigned)(unsigned long long)(uintptr_t)p;   // LDS generic addr: low 32 bits = offset
}

// Load a 16x32 bf16 fragment from an LDS row-major tile.
// A layout (ISA 7.12.2): lane r holds row r, K chunks {0..7,16..23}; lane r+16 the
// complementary chunks. B fragments use the same pattern on B^T rows, so weights
// are staged transposed. Requires 16B-aligned per-lane addresses.
__device__ __forceinline__ v16bf load_frag16x32(const bf16_t* __restrict__ p0,
                                                int stride, int lane) {
    int r  = lane & 15;
    int hi = (lane >> 4) & 1;
    const bf16_t* p = p0 + r * stride + hi * 8;
    v8bf lo = *(const v8bf*)(p);        // K = hi*8 .. hi*8+7
    v8bf hh = *(const v8bf*)(p + 16);   // K = 16 + hi*8 ..
    v16bf f;
#pragma unroll
    for (int e = 0; e < 8; ++e) { f[e] = lo[e]; f[e + 8] = hh[e]; }
    return f;
}

// Stage src (rows x cols, f32, row-major, global) transposed into LDS bf16
// dst[cols][dstride]: dst[c][r] = src[r][c]. Coalesced global reads.
__device__ __forceinline__ void stage_transpose(const float* __restrict__ src,
                                                bf16_t* __restrict__ dst,
                                                int rows, int cols, int dstride,
                                                int t, int nt) {
    int total = rows * cols;
    for (int idx = t; idx < total; idx += nt) {
        int r = idx / cols, c = idx % cols;
        dst[c * dstride + r] = (bf16_t)src[idx];
    }
}

// ---------------- kernel 0: xs init (xs lives in d_out) ----------------
__global__ void __launch_bounds__(256)
init_xs_kernel(const float* __restrict__ emb_e, float* __restrict__ xs) {
    int idx = blockIdx.x * 256 + threadIdx.x;   // over BS*NE*DD
    int i = (idx >> 7) & (NE - 1);
    int d = idx & (DD - 1);
    xs[idx] = emb_e[i * DD + d];
}

// ---------------- kernel 1: zs = xs @ eiW (electron rows of zs_all) ----------
__global__ void __launch_bounds__(256)
embed_in_kernel(const float* __restrict__ xs, const float* __restrict__ eiW,
                float* __restrict__ zs_all) {
    extern __shared__ char smem[];
    bf16_t* wT = (bf16_t*)smem;               // 128 x 136 bf16 (eiW^T) 34816B
    bf16_t* aT = (bf16_t*)(smem + 34816);     //  64 x 136 bf16         17408B
    int t = threadIdx.x;
    int m0 = blockIdx.x * 64;                 // global row = b*NE + j

    stage_transpose(eiW, wT, DD, KD, 136, t, 256);
    for (int idx = t; idx < 64 * DD; idx += 256) {
        int r = idx >> 7, k = idx & 127;
        aT[r * 136 + k] = (bf16_t)xs[(m0 + r) * DD + k];
    }
    __syncthreads();

    int wv = t >> 5, lane = t & 31;
    int n0 = wv * 16;
    int ln = lane & 15, hi = (lane >> 4) & 1;
#pragma unroll
    for (int mt = 0; mt < 4; ++mt) {
        v8f acc = V8F_ZERO;
#pragma unroll
        for (int kc = 0; kc < 4; ++kc) {
            v16bf af = load_frag16x32(aT + (mt * 16) * 136 + kc * 32, 136, lane);
            v16bf bf = load_frag16x32(wT + n0 * 136 + kc * 32, 136, lane);
            acc = wmma_bf16(af, bf, acc);
        }
        int n = n0 + ln;
#pragma unroll
        for (int v = 0; v < 8; ++v) {
            int m = m0 + mt * 16 + hi * 8 + v;       // = b*NE + j
            int b = m >> 6, j = m & (NE - 1);
            zs_all[((size_t)b * NP + j) * KD + n] = acc[v];
        }
    }
}

// ---------------- kernel 2: nuclei rows of zs_all ----------------
__global__ void __launch_bounds__(256)
copy_nuc_kernel(const float* __restrict__ emb_n, float* __restrict__ zs_all) {
    int idx = blockIdx.x * 256 + threadIdx.x;   // over BS*NN*KD
    int b = idx >> 11;
    int r = (idx >> 7) & (NN - 1);
    int k = idx & (KD - 1);
    zs_all[((size_t)b * NP + NE + r) * KD + k] = emb_n[r * KD + k];
}

// ---------------- kernel 3: fused filter MLP + mask + cfconv aggregation ------
// One WG = (b, group of 4 electrons). zs block and basis tiles arrive via
// GLOBAL_LOAD_ASYNC_TO_LDS_B128 (double-buffered basis prefetch across electrons).
__global__ void __launch_bounds__(256)
cfconv_kernel(const float* __restrict__ basis, const float* __restrict__ zs_all,
              const float* __restrict__ kW1, const float* __restrict__ kb1,
              const float* __restrict__ kW2, const float* __restrict__ kb2,
              float* __restrict__ zagg) {
    extern __shared__ char smem[];
    float*  zsb = (float*)smem;                    // 80 x 132 f32   42240B
    bf16_t* w2t = (bf16_t*)(smem + 42240);         // 128 x 136 bf16 34816B
    bf16_t* w1t = (bf16_t*)(smem + 77056);         // 128 x 40  bf16 10240B
    bf16_t* hb  = (bf16_t*)(smem + 87296);         //  80 x 136 bf16 21760B
    bf16_t* ab  = (bf16_t*)(smem + 109056);        //  80 x 40  bf16  6400B
    float*  lb1 = (float*)(smem + 115456);         // 128 f32
    float*  lb2 = (float*)(smem + 115968);         // 128 f32
    float*  acl = (float*)(smem + 116480);         // 128 f32 accumulators
    float*  braw0 = (float*)(smem + 116992);       // raw basis buf 0  10240B
    float*  braw1 = (float*)(smem + 127232);       // raw basis buf 1  10240B
    // total 137472B

    int t = threadIdx.x;
    int g = blockIdx.x;          // 4096 = BS * (NE/4)
    int b  = g >> 4;
    int i0 = (g & 15) * 4;
    int wv = t >> 5, lane = t & 31;

    // --- async: zs block (f32, consumed as f32) straight into padded layout ---
    {
        unsigned zb = lds_off(zsb);
        for (int c = t; c < NP * 32; c += 256) {          // 2560 x 16B chunks
            int j = c >> 5, q = c & 31;
            async_copy_b128(zb + (unsigned)((j * 132 + q * 4) * 4),
                            zs_all + ((size_t)b * NP + j) * KD + q * 4);
        }
    }
    // --- async: prefetch basis tile for electron i0 into braw0 ---
    {
        const float* bp = basis + ((size_t)(b * NE + i0)) * NP * NB;
        unsigned ba = lds_off(braw0);
        for (int c = t; c < NP * NB / 4; c += 256)        // 640 x 16B chunks
            async_copy_b128(ba + (unsigned)c * 16u, bp + c * 4);
    }

    // --- VALU staging of weights overlaps with async traffic ---
    stage_transpose(kW1, w1t, NB, KD, 40, t, 256);
    stage_transpose(kW2, w2t, KD, KD, 136, t, 256);
    if (t < KD) { lb1[t] = kb1[t]; lb2[t] = kb2[t]; }

    int k0 = wv * 16;
    int ln = lane & 15, hi = (lane >> 4) & 1;

    for (int ii = 0; ii < 4; ++ii) {
        int i = i0 + ii;
        float* bcur = (ii & 1) ? braw1 : braw0;
        float* bnxt = (ii & 1) ? braw0 : braw1;

        // prefetch next electron's basis while we compute this one
        if (ii < 3) {
            const float* bp = basis + ((size_t)(b * NE + i + 1)) * NP * NB;
            unsigned ba = lds_off(bnxt);
            for (int c = t; c < NP * NB / 4; c += 256)
                async_copy_b128(ba + (unsigned)c * 16u, bp + c * 4);
            // drain everything older than the prefetch just issued
            // (waves 0-3 issued 3 chunks of it, waves 4-7 issued 2)
            if (wv < 4) asm volatile("s_wait_asynccnt 3" ::: "memory");
            else        asm volatile("s_wait_asynccnt 2" ::: "memory");
        } else {
            asm volatile("s_wait_asynccnt 0" ::: "memory");
        }
        __syncthreads();   // current basis + zs visible to all waves

        // LDS-resident f32 -> bf16 conversion of current basis tile; reset acc
        for (int idx = t; idx < NP * NB; idx += 256)
            ab[(idx >> 5) * 40 + (idx & 31)] = (bf16_t)bcur[idx];
        if (t < KD) acl[t] = 0.f;
        __syncthreads();

        // GEMM1: 80x32 @ 32x128 -> ssp -> hb (bf16)
#pragma unroll
        for (int mt = 0; mt < 5; ++mt) {
            v16bf af = load_frag16x32(ab + mt * 16 * 40, 40, lane);
            v16bf bf = load_frag16x32(w1t + k0 * 40, 40, lane);
            v8f d = V8F_ZERO;
            d = wmma_bf16(af, bf, d);
            float bias = lb1[k0 + ln];
#pragma unroll
            for (int v = 0; v < 8; ++v) {
                int m = mt * 16 + hi * 8 + v;
                hb[m * 136 + k0 + ln] = (bf16_t)sspf(d[v] + bias);
            }
        }
        __syncthreads();

        // GEMM2 + masked weighted j-reduction
        float part = 0.f;
        float bias2 = lb2[k0 + ln];
#pragma unroll
        for (int mt = 0; mt < 5; ++mt) {
            v8f d = V8F_ZERO;
#pragma unroll
            for (int kc = 0; kc < 4; ++kc) {
                v16bf af = load_frag16x32(hb + mt * 16 * 136 + kc * 32, 136, lane);
                v16bf bf = load_frag16x32(w2t + k0 * 136 + kc * 32, 136, lane);
                d = wmma_bf16(af, bf, d);
            }
#pragma unroll
            for (int v = 0; v < 8; ++v) {
                int j = mt * 16 + hi * 8 + v;
                float z = (j == i) ? 0.f : zsb[j * 132 + (k0 + ln)];  // ZeroDiag mask
                part += (d[v] + bias2) * z;
            }
        }
        atomicAdd(&acl[k0 + ln], part);   // ds_add_f32 across waves / lane halves
        __syncthreads();
        if (t < KD) zagg[((size_t)(b * NE + i)) * KD + t] = acl[t];
    }
}

// ---------------- kernel 4: output MLP + residual (in-place on xs) ----------
__global__ void __launch_bounds__(256)
outmlp_kernel(const float* __restrict__ zin, const float* __restrict__ eoW1,
              const float* __restrict__ eob1, const float* __restrict__ eoW2,
              const float* __restrict__ eob2, float* __restrict__ xs) {
    extern __shared__ char smem[];
    bf16_t* w1t = (bf16_t*)smem;                   // 128 x 136  34816B
    bf16_t* w2t = (bf16_t*)(smem + 34816);         // 128 x 136  34816B
    bf16_t* zt  = (bf16_t*)(smem + 69632);         //  64 x 136  17408B
    bf16_t* yt  = (bf16_t*)(smem + 87040);         //  64 x 136  17408B
    float*  lb1 = (float*)(smem + 104448);
    float*  lb2 = (float*)(smem + 104960);

    int t = threadIdx.x;
    int m0 = blockIdx.x * 64;

    stage_transpose(eoW1, w1t, KD, DD, 136, t, 256);
    stage_transpose(eoW2, w2t, DD, DD, 136, t, 256);
    for (int idx = t; idx < 64 * KD; idx += 256) {
        int r = idx >> 7, k = idx & 127;
        zt[r * 136 + k] = (bf16_t)zin[(m0 + r) * KD + k];
    }
    if (t < DD) { lb1[t] = eob1[t]; lb2[t] = eob2[t]; }
    __syncthreads();

    int wv = t >> 5, lane = t & 31;
    int n0 = wv * 16;
    int ln = lane & 15, hi = (lane >> 4) & 1;

    // GEMM1 + ssp -> yt
#pragma unroll
    for (int mt = 0; mt < 4; ++mt) {
        v8f acc = V8F_ZERO;
#pragma unroll
        for (int kc = 0; kc < 4; ++kc) {
            v16bf af = load_frag16x32(zt + mt * 16 * 136 + kc * 32, 136, lane);
            v16bf bf = load_frag16x32(w1t + n0 * 136 + kc * 32, 136, lane);
            acc = wmma_bf16(af, bf, acc);
        }
        float bias = lb1[n0 + ln];
#pragma unroll
        for (int v = 0; v < 8; ++v) {
            int m = mt * 16 + hi * 8 + v;
            yt[m * 136 + n0 + ln] = (bf16_t)sspf(acc[v] + bias);
        }
    }
    __syncthreads();

    // GEMM2 + bias + residual
#pragma unroll
    for (int mt = 0; mt < 4; ++mt) {
        v8f acc = V8F_ZERO;
#pragma unroll
        for (int kc = 0; kc < 4; ++kc) {
            v16bf af = load_frag16x32(yt + mt * 16 * 136 + kc * 32, 136, lane);
            v16bf bf = load_frag16x32(w2t + n0 * 136 + kc * 32, 136, lane);
            acc = wmma_bf16(af, bf, acc);
        }
        float bias = lb2[n0 + ln];
#pragma unroll
        for (int v = 0; v < 8; ++v) {
            int m = mt * 16 + hi * 8 + v;
            float* p = xs + (size_t)(m0 + m) * DD + (n0 + ln);
            *p = *p + acc[v] + bias;
        }
    }
}

// ---------------- host side ----------------
extern "C" void kernel_launch(void* const* d_in, const int* in_sizes, int n_in,
                              void* d_out, int out_size, void* d_ws, size_t ws_size,
                              hipStream_t stream) {
    const float* basis = (const float*)d_in[0];
    const float* emb_e = (const float*)d_in[1];
    const float* emb_n = (const float*)d_in[2];
    const float* kW1   = (const float*)d_in[3];
    const float* kb1   = (const float*)d_in[4];
    const float* kW2   = (const float*)d_in[5];
    const float* kb2   = (const float*)d_in[6];
    const float* eiW   = (const float*)d_in[7];
    const float* eoW1  = (const float*)d_in[8];
    const float* eob1  = (const float*)d_in[9];
    const float* eoW2  = (const float*)d_in[10];
    const float* eob2  = (const float*)d_in[11];

    float* xs     = (float*)d_out;                       // (BS, NE, DD), in place
    float* zs_all = (float*)d_ws;                        // (BS, NP, KD) = 10.5 MB
    float* zagg   = zs_all + (size_t)BS * NP * KD;       // (BS, NE, KD) =  8.4 MB

    init_xs_kernel<<<(BS * NE * DD) / 256, 256, 0, stream>>>(emb_e, xs);

    for (int l = 0; l < NL; ++l) {
        embed_in_kernel<<<BS * NE / 64, 256, 52224, stream>>>(
            xs, eiW + (size_t)l * DD * KD, zs_all);
        copy_nuc_kernel<<<(BS * NN * KD) / 256, 256, 0, stream>>>(emb_n, zs_all);
        cfconv_kernel<<<BS * NE / 4, 256, 137472, stream>>>(
            basis, zs_all,
            kW1 + (size_t)l * NB * KD, kb1 + (size_t)l * KD,
            kW2 + (size_t)l * KD * KD, kb2 + (size_t)l * KD, zagg);
        outmlp_kernel<<<BS * NE / 64, 256, 105472, stream>>>(
            zagg, eoW1 + (size_t)l * KD * DD, eob1 + (size_t)l * DD,
            eoW2 + (size_t)l * DD * DD, eob2 + (size_t)l * DD, xs);
    }
}
